// MultiheadAttention_72524817760404
// MI455X (gfx1250) — compile-verified, matrix operand order checked by reading, not run on hardware
//
#include <hip/hip_runtime.h>

// ---------------------------------------------------------------------------
// MHA forward for B=4, S=2048, E=2048, H=16, D=128 on gfx1250 (CDNA5).
// All matmuls via v_wmma_f32_16x16x32_bf16 (fp32 accumulate).
// Global->LDS staging uses GLOBAL_LOAD_ASYNC_TO_LDS_B128 (ASYNCcnt) with a
// double-buffered k-loop in the GEMMs.
// ---------------------------------------------------------------------------

typedef __bf16 bf16;
typedef __attribute__((ext_vector_type(16))) __bf16 v16bf;
typedef __attribute__((ext_vector_type(8)))  __bf16 v8bf;
typedef __attribute__((ext_vector_type(4)))  __bf16 v4bf;
typedef __attribute__((ext_vector_type(8)))  float  v8f;
typedef __attribute__((ext_vector_type(4)))  int    v4i;

// Pointer-to-v4i in global (AS1) / LDS (AS3) address spaces, matching the
// async-load builtin's parameter types (revealed by the clang diagnostic).
typedef __attribute__((address_space(1))) v4i* gv4i_p;
typedef __attribute__((address_space(3))) v4i* lv4i_p;

#if defined(__has_builtin)
#  if __has_builtin(__builtin_amdgcn_global_load_async_to_lds_b128)
#    define HAS_ASYNC_LDS 1
#  endif
#endif
#ifndef HAS_ASYNC_LDS
#  define HAS_ASYNC_LDS 0
#endif

#define WMMA_BF16(a, b, c) \
  __builtin_amdgcn_wmma_f32_16x16x32_bf16(false, (a), false, (b), (short)0, (c), false, false)

// 16-byte global -> LDS copy, async when the toolchain exposes the builtin.
static __device__ __forceinline__ void cp16_async(bf16* lds_dst, const bf16* gsrc) {
#if HAS_ASYNC_LDS
  __builtin_amdgcn_global_load_async_to_lds_b128((gv4i_p)gsrc, (lv4i_p)lds_dst, 0, 0);
#else
  *(v8bf*)lds_dst = *(const v8bf*)gsrc;
#endif
}

template <int N>
static __device__ __forceinline__ void wait_async() {
#if defined(__has_builtin) && __has_builtin(__builtin_amdgcn_s_wait_asynccnt)
  __builtin_amdgcn_s_wait_asynccnt(N);
#else
  asm volatile("s_wait_asynccnt %0" ::"i"(N) : "memory");
#endif
}

// Load one 16x32 bf16 fragment (A-layout / row-major-B layout) from a
// row-major [16][ld] tile.  ISA 7.12.2: lane = (r = lane&15, kh = lane>>4);
// VGPR0..3 hold K = kh*8 .. +7, VGPR4..7 hold K = 16+kh*8 .. +7  ->
// two contiguous b128 loads per lane.
static __device__ __forceinline__ v16bf frag_ld(const bf16* base, int ld) {
  const int lane = threadIdx.x & 31;
  const int r = lane & 15, kh = lane >> 4;
  const bf16* p = base + r * ld + kh * 8;
  v8bf lo = *(const v8bf*)(p);
  v8bf hi = *(const v8bf*)(p + 16);
  v16bf f;
#pragma unroll
  for (int i = 0; i < 8; ++i) { f[i] = lo[i]; f[i + 8] = hi[i]; }
  return f;
}

// ---------------------------------------------------------------------------
// fp32 -> bf16 conversion (vectorized x4)
// ---------------------------------------------------------------------------
__global__ __launch_bounds__(256) void cvt_f32_bf16(const float* __restrict__ in,
                                                    bf16* __restrict__ out, int n) {
  int i = (blockIdx.x * 256 + threadIdx.x) * 4;
  if (i + 3 < n) {
    float4 f = *(const float4*)(in + i);
    v4bf o; o[0] = (bf16)f.x; o[1] = (bf16)f.y; o[2] = (bf16)f.z; o[3] = (bf16)f.w;
    *(v4bf*)(out + i) = o;
  } else {
    for (; i < n; ++i) out[i] = (bf16)in[i];
  }
}

// ---------------------------------------------------------------------------
// Tiled GEMM:  Y[m,n] = sum_k A[m,k] * W[n,k] + bias[n]      (A @ W^T + b)
// BM=BN=128, BK=32, double-buffered LDS, async global->LDS staging.
// 8 waves in a 4x2 grid: each wave owns 32x64 (2 A-frags x 4 B-frags,
// 8 WMMAs per 12 ds_load_b128).
// ---------------------------------------------------------------------------
template <bool SPLIT_HEADS, bool OUT_BF16>
__global__ __launch_bounds__(256) void gemm_bf16_wmma(
    const bf16* __restrict__ A,   // [M][K]
    const bf16* __restrict__ W,   // [N][K]
    const float* __restrict__ bias,
    void* __restrict__ Out, int M, int N, int K) {
  constexpr int LDK = 40;                 // 32 + 8 pad, rows stay 16B aligned
  __shared__ bf16 As[2][128 * LDK];
  __shared__ bf16 Bs[2][128 * LDK];

  const int tid = threadIdx.x;
  const int wave = tid >> 5, lane = tid & 31;
  const int wr = wave >> 1, wc = wave & 1;          // 4x2 wave grid
  const int m0 = blockIdx.y * 128, n0 = blockIdx.x * 128;
  const int lr = tid >> 2;                 // 0..63
  const int lc = (tid & 3) * 8;            // 0,8,16,24

  v8f acc[2][4] = {};
  const int nk = K >> 5;

  auto stage = [&](int kk, int b) {        // 4 async b128 per thread
    const size_t ko = (size_t)kk * 32 + lc;
    cp16_async(&As[b][lr * LDK + lc],        A + (size_t)(m0 + lr) * K + ko);
    cp16_async(&As[b][(lr + 64) * LDK + lc], A + (size_t)(m0 + lr + 64) * K + ko);
    cp16_async(&Bs[b][lr * LDK + lc],        W + (size_t)(n0 + lr) * K + ko);
    cp16_async(&Bs[b][(lr + 64) * LDK + lc], W + (size_t)(n0 + lr + 64) * K + ko);
  };

  stage(0, 0);
  for (int kk = 0; kk < nk; ++kk) {
    const int cur = kk & 1;
    if (kk + 1 < nk) { stage(kk + 1, cur ^ 1); wait_async<4>(); }  // in-order:
    else             { wait_async<0>(); }       // <=4 left => tile kk resident
    __syncthreads();

    const bf16* as = As[cur];
    const bf16* bs = Bs[cur];
    v16bf a0 = frag_ld(as + (wr * 32) * LDK, LDK);
    v16bf a1 = frag_ld(as + (wr * 32 + 16) * LDK, LDK);
    v16bf b0 = frag_ld(bs + (wc * 64) * LDK, LDK);
    v16bf b1 = frag_ld(bs + (wc * 64 + 16) * LDK, LDK);
    v16bf b2 = frag_ld(bs + (wc * 64 + 32) * LDK, LDK);
    v16bf b3 = frag_ld(bs + (wc * 64 + 48) * LDK, LDK);
    acc[0][0] = WMMA_BF16(a0, b0, acc[0][0]);
    acc[0][1] = WMMA_BF16(a0, b1, acc[0][1]);
    acc[0][2] = WMMA_BF16(a0, b2, acc[0][2]);
    acc[0][3] = WMMA_BF16(a0, b3, acc[0][3]);
    acc[1][0] = WMMA_BF16(a1, b0, acc[1][0]);
    acc[1][1] = WMMA_BF16(a1, b1, acc[1][1]);
    acc[1][2] = WMMA_BF16(a1, b2, acc[1][2]);
    acc[1][3] = WMMA_BF16(a1, b3, acc[1][3]);
    __syncthreads();   // readers done before buffer is re-staged next iter
  }

  // ---- epilogue: C/D layout is (row = j + 8*(lane>>4), col = lane&15) ----
  const int nl = lane & 15, kh = lane >> 4;
#pragma unroll
  for (int a = 0; a < 2; ++a) {
#pragma unroll
    for (int b = 0; b < 4; ++b) {
      const int ncol = n0 + wc * 64 + b * 16 + nl;
      const float bv = bias[ncol];
#pragma unroll
      for (int j = 0; j < 8; ++j) {
        const int mrow = m0 + wr * 32 + a * 16 + j + 8 * kh;
        const float val = acc[a][b][j] + bv;
        if (OUT_BF16) {
          bf16* o = (bf16*)Out;
          if (SPLIT_HEADS) {  // mrow = bb*2048 + s ; ncol = h*128 + d -> [B,H,S,D]
            const int bb = mrow >> 11, s = mrow & 2047;
            const int h = ncol >> 7,   d = ncol & 127;
            o[((size_t)(bb * 16 + h) * 2048 + s) * 128 + d] = (bf16)val;
          } else {
            o[(size_t)mrow * N + ncol] = (bf16)val;
          }
        } else {
          ((float*)Out)[(size_t)mrow * N + ncol] = val;
        }
      }
    }
  }
}

// ---------------------------------------------------------------------------
// Causal flash attention, one workgroup per (b*H+h, 128-query block).
// 8 waves; wave owns 16 query rows.  Q frags resident in registers.
// K block async-staged into LDS [key][d]; V staged transposed [d][key];
// P relayout C->A via per-wave LDS scratch.
// ---------------------------------------------------------------------------
__global__ __launch_bounds__(256) void attn_wmma(
    const bf16* __restrict__ Q,   // [B*H][S][D]
    const bf16* __restrict__ Km,  // [B*H][S][D]
    const bf16* __restrict__ V,   // [B*H][S][D]
    bf16* __restrict__ Ctx) {     // [B][S][E] merged heads
  constexpr int S = 2048, D = 128, LD = 136;  // 128 + 8 pad
  extern __shared__ char smem[];
  bf16* Ks = (bf16*)smem;            // 128 x LD
  bf16* Vt = Ks + 128 * LD;          // 128 x LD  (transposed V: [d][key])
  bf16* Ps = Vt + 128 * LD;          // 8 waves x 16 x LD

  const int tid = threadIdx.x;
  const int wave = tid >> 5, lane = tid & 31;
  const int nl = lane & 15, kh = lane >> 4;
  const int qb = blockIdx.x;         // query block (16 blocks of 128)
  const int bh = blockIdx.y;         // b*H + h     (64)
  const bf16* Qh = Q  + (size_t)bh * S * D;
  const bf16* Kh = Km + (size_t)bh * S * D;
  const bf16* Vh = V  + (size_t)bh * S * D;
  const int q0 = qb * 128 + wave * 16;

  v16bf qf[4];
#pragma unroll
  for (int ks = 0; ks < 4; ++ks) qf[ks] = frag_ld(Qh + (size_t)q0 * D + ks * 32, D);

  v8f acco[8] = {};
  float mrow[8], lrow[8];
#pragma unroll
  for (int j = 0; j < 8; ++j) { mrow[j] = -1e30f; lrow[j] = 0.f; }

  const float scale = 0.08838834764831845f;   // 1/sqrt(128)
  bf16* ps = Ps + wave * 16 * LD;

  for (int kb = 0; kb <= qb; ++kb) {
    __syncthreads();   // previous iteration's readers are done
    // --- async-stage K block [key][d]: 8 b128 per thread ---
    {
      const int r = tid >> 1, c0 = (tid & 1) * 8;
      const bf16* krow = Kh + (size_t)(kb * 128 + r) * D;
#pragma unroll
      for (int c = c0; c < D; c += 16)
        cp16_async(Ks + r * LD + c, krow + c);
    }
    // --- stage V transposed: Vt[d][key] (needs VGPR round trip) ---
#pragma unroll 4
    for (int i = 0; i < 16; ++i) {
      const int chunk = i * 256 + tid;        // 4096 chunks of 4 elems
      const int key = chunk >> 5, d4 = (chunk & 31) * 4;
      v4bf vv = *(const v4bf*)(Vh + (size_t)(kb * 128 + key) * D + d4);
#pragma unroll
      for (int t = 0; t < 4; ++t) Vt[(d4 + t) * LD + key] = vv[t];
    }
    wait_async<0>();
    __syncthreads();

    // --- scores: 16 queries x 128 keys, 8 tiles of 16x16 ---
    v8f sc[8];
#pragma unroll
    for (int kt = 0; kt < 8; ++kt) {
      v8f s = {};
#pragma unroll
      for (int ks = 0; ks < 4; ++ks) {
        v16bf bfrag = frag_ld(Ks + kt * 16 * LD + ks * 32, LD);
        s = WMMA_BF16(qf[ks], bfrag, s);
      }
      sc[kt] = s;
    }

    // --- scale + causal mask (no-op for kb < qb) ---
#pragma unroll
    for (int kt = 0; kt < 8; ++kt) {
      const int key = kb * 128 + kt * 16 + nl;
#pragma unroll
      for (int j = 0; j < 8; ++j) {
        const int qr = qb * 128 + wave * 16 + j + 8 * kh;
        float s = sc[kt][j] * scale;
        if (key > qr) s = -1e30f;
        sc[kt][j] = s;
      }
    }

    // --- online softmax update (reduce across 16 n-lanes per half) ---
#pragma unroll
    for (int j = 0; j < 8; ++j) {
      float mx = -1e30f;
#pragma unroll
      for (int kt = 0; kt < 8; ++kt) mx = fmaxf(mx, sc[kt][j]);
#pragma unroll
      for (int off = 1; off < 16; off <<= 1) mx = fmaxf(mx, __shfl_xor(mx, off, 32));
      const float mn = fmaxf(mrow[j], mx);
      const float corr = __expf(mrow[j] - mn);
      float sum = 0.f;
#pragma unroll
      for (int kt = 0; kt < 8; ++kt) {
        const float p = __expf(sc[kt][j] - mn);
        sc[kt][j] = p; sum += p;
      }
#pragma unroll
      for (int off = 1; off < 16; off <<= 1) sum += __shfl_xor(sum, off, 32);
      lrow[j] = lrow[j] * corr + sum;
      mrow[j] = mn;
#pragma unroll
      for (int dt = 0; dt < 8; ++dt) acco[dt][j] *= corr;
    }

    // --- P: C-layout -> A-layout via per-wave LDS staging (bf16) ---
#pragma unroll
    for (int kt = 0; kt < 8; ++kt)
#pragma unroll
      for (int j = 0; j < 8; ++j)
        ps[(j + 8 * kh) * LD + kt * 16 + nl] = (bf16)sc[kt][j];

    // --- acco += P @ V  (k-dim = 128 keys, n = 128 d-cols) ---
#pragma unroll
    for (int ks = 0; ks < 4; ++ks) {
      v16bf pf = frag_ld(ps + ks * 32, LD);
#pragma unroll
      for (int dt = 0; dt < 8; ++dt) {
        v16bf vf = frag_ld(Vt + dt * 16 * LD + ks * 32, LD);
        acco[dt] = WMMA_BF16(pf, vf, acco[dt]);
      }
    }
  }

  // --- epilogue: ctx[b, s, h*128 + d] = acco / l ---
  const int b = bh >> 4, h = bh & 15;
  bf16* outp = Ctx + (size_t)b * 2048 * 2048 + h * 128;
#pragma unroll
  for (int dt = 0; dt < 8; ++dt) {
#pragma unroll
    for (int j = 0; j < 8; ++j) {
      const int qr = qb * 128 + wave * 16 + j + 8 * kh;
      outp[(size_t)qr * 2048 + dt * 16 + nl] = (bf16)(acco[dt][j] / lrow[j]);
    }
  }
}

// ---------------------------------------------------------------------------
// Launcher
// ---------------------------------------------------------------------------
extern "C" void kernel_launch(void* const* d_in, const int* in_sizes, int n_in,
                              void* d_out, int out_size, void* d_ws, size_t ws_size,
                              hipStream_t stream) {
  const float* x  = (const float*)d_in[0];
  // d_in[1] = causal_mask (implicit in the kernel)
  const float* wq = (const float*)d_in[2]; const float* bq = (const float*)d_in[3];
  const float* wk = (const float*)d_in[4]; const float* bk = (const float*)d_in[5];
  const float* wv = (const float*)d_in[6]; const float* bv = (const float*)d_in[7];
  const float* wo = (const float*)d_in[8]; const float* bo = (const float*)d_in[9];
  float* out = (float*)d_out;

  const size_t NX = (size_t)8192 * 2048;   // B*S x E
  const size_t NW = (size_t)2048 * 2048;

  bf16* xb   = (bf16*)d_ws;
  bf16* wqb  = xb   + NX;
  bf16* wkb  = wqb  + NW;
  bf16* wvb  = wkb  + NW;
  bf16* wob  = wvb  + NW;
  bf16* qbuf = wob  + NW;   // [B,H,S,D]
  bf16* kbuf = qbuf + NX;
  bf16* vbuf = kbuf + NX;
  bf16* ctxb = vbuf + NX;   // [B,S,E]

  // 1) precision conversion
  cvt_f32_bf16<<<(int)(NX / 4 + 255) / 256, 256, 0, stream>>>(x,  xb,  (int)NX);
  cvt_f32_bf16<<<(int)(NW / 4 + 255) / 256, 256, 0, stream>>>(wq, wqb, (int)NW);
  cvt_f32_bf16<<<(int)(NW / 4 + 255) / 256, 256, 0, stream>>>(wk, wkb, (int)NW);
  cvt_f32_bf16<<<(int)(NW / 4 + 255) / 256, 256, 0, stream>>>(wv, wvb, (int)NW);
  cvt_f32_bf16<<<(int)(NW / 4 + 255) / 256, 256, 0, stream>>>(wo, wob, (int)NW);

  // 2) QKV projections (split-head bf16 output)
  dim3 gg(2048 / 128, 8192 / 128);
  gemm_bf16_wmma<true,  true ><<<gg, 256, 0, stream>>>(xb, wqb, bq, qbuf, 8192, 2048, 2048);
  gemm_bf16_wmma<true,  true ><<<gg, 256, 0, stream>>>(xb, wkb, bk, kbuf, 8192, 2048, 2048);
  gemm_bf16_wmma<true,  true ><<<gg, 256, 0, stream>>>(xb, wvb, bv, vbuf, 8192, 2048, 2048);

  // 3) causal flash attention
  const int attn_lds = (128 * 136 + 128 * 136 + 8 * 16 * 136) * (int)sizeof(bf16);
  attn_wmma<<<dim3(2048 / 128, 64), 256, attn_lds, stream>>>(qbuf, kbuf, vbuf, ctxb);

  // 4) output projection (fp32 + bias to d_out)
  gemm_bf16_wmma<false, false><<<gg, 256, 0, stream>>>(ctxb, wob, bo, out, 8192, 2048, 2048);
}
